// SelfAttention_2422361555212
// MI455X (gfx1250) — matrix operand order repaired and checked by reading, not compiled, not run on hardware
//
#include <hip/hip_runtime.h>
#include <hip/hip_bf16.h>

// ---------------------------------------------------------------------------
// Single-head self-attention, B=4, N=2048, E=1024, fp32 in/out.
//   (0) convert x, W to f16 (one elementwise pass)
//   (1) QKV projection GEMM (f16 WMMA, f32 accum), 32x64 tiles/wave
//   (2) scores = QK^T * scale + row softmax -> P f16 (Q staged via TDM->LDS)
//   (3) out = P @ V (via Vt), 32x64 tiles/wave
// Workspace (80MB):
//   [0,16M)   Qh  f16 [B][N][E]
//   [16M,32M) Kh  f16 [B][N][E]
//   [32M,48M) Vt  f16 [B][E][N]
//   [48M,80M) P   f16 [B][N][N]   (aliased: Xh @48M, Wh @64M, dead before P)
// All GEMM loops use ONE induction pointer per operand; other fragment rows
// are reached via constant byte offsets (fits 24-bit VMEM IOFFSET) to keep
// address pressure at 2 VGPR pairs and avoid spills around the 64-VGPR accum.
// ---------------------------------------------------------------------------

typedef __attribute__((ext_vector_type(16))) _Float16 v16h;
typedef __attribute__((ext_vector_type(8)))  _Float16 v8h;
typedef __attribute__((ext_vector_type(8)))  float    v8f;
typedef __attribute__((ext_vector_type(4)))  float    v4f;
typedef __attribute__((ext_vector_type(4)))  unsigned int u32x4;
typedef __attribute__((ext_vector_type(8)))  int      i32x8;
typedef __attribute__((ext_vector_type(4)))  int      i32x4;
typedef __attribute__((ext_vector_type(4)))  unsigned int uint4v;

#define ATT_B 4
#define ATT_N 2048
#define ATT_E 1024
#define ATT_F 3072
#define ATT_SCALE 0.03125f   // E^-0.5

// A 16x32 f16 fragment from two contiguous 8-half chunks (ISA 7.12.2):
// lane = sub*16+r holds row M=r, elems 0..7 = K(sub*16+i), 8..15 = K(sub*16+8+i).
// Caller's pointer is pre-offset by sub*16; lo at +0, hi at +8.
__device__ __forceinline__ v16h make_a(v8h lo, v8h hi) {
  v16h r;
#pragma unroll
  for (int i = 0; i < 8; ++i) { r[i] = lo[i]; r[i + 8] = hi[i]; }
  return r;
}

__device__ __forceinline__ v8f wmma_f16(v16h a, v16h b, v8f c) {
  return __builtin_amdgcn_wmma_f32_16x16x32_f16(false, a, false, b,
                                                (short)0, c, false, false);
}

// ---------------------------------------------------------------------------
// Kernel 0: fp32 -> f16 convert (8 elems/thread, b128 in / b128 out)
// ---------------------------------------------------------------------------
__global__ void cvt_f16_kernel(const float* __restrict__ src,
                               _Float16* __restrict__ dst) {
  const size_t i = ((size_t)blockIdx.x * blockDim.x + threadIdx.x) * 8;
  v4f a = *(const v4f*)(src + i);
  v4f b = *(const v4f*)(src + i + 4);
  v8h o;
#pragma unroll
  for (int j = 0; j < 4; ++j) { o[j] = (_Float16)a[j]; o[j + 4] = (_Float16)b[j]; }
  *(v8h*)(dst + i) = o;
}

// ---------------------------------------------------------------------------
// Kernel 1: qkv = x @ W^T  (M=8192, F=3072, K=1024), 32x64 tile per wave
// ---------------------------------------------------------------------------
__global__ void qkv_proj_kernel(const _Float16* __restrict__ Xh,
                                const _Float16* __restrict__ Wh,
                                _Float16* __restrict__ Qh,
                                _Float16* __restrict__ Kh,
                                _Float16* __restrict__ Vt) {
  const int lane = threadIdx.x & 31;
  const int wid  = threadIdx.x >> 5;
  const int sub  = lane >> 4;
  const int r16  = lane & 15;

  const int tile  = blockIdx.x * 4 + wid;       // 256 * 48 tiles
  const int tileM = tile / (ATT_F / 64);
  const int tileF = tile % (ATT_F / 64);
  const int m0 = tileM * 32;
  const int f0 = tileF * 64;

  // single induction pointer per operand; all other rows via constant offsets
  const _Float16* pa = Xh + (size_t)(m0 + r16) * ATT_E + sub * 16;
  const _Float16* pb = Wh + (size_t)(f0 + r16) * ATT_E + sub * 16;

  v8f acc[2][4] = {};
#pragma unroll 2
  for (int k0 = 0; k0 < ATT_E; k0 += 32) {
    v16h a0 = make_a(*(const v8h*)(pa),
                     *(const v8h*)(pa + 8));
    v16h a1 = make_a(*(const v8h*)(pa + 16 * ATT_E),
                     *(const v8h*)(pa + 16 * ATT_E + 8));
    v16h bf[4];
#pragma unroll
    for (int t = 0; t < 4; ++t)
      bf[t] = *(const v16h*)(pb + t * 16 * ATT_E);
#pragma unroll
    for (int t = 0; t < 4; ++t) {
      acc[0][t] = wmma_f16(a0, bf[t], acc[0][t]);
      acc[1][t] = wmma_f16(a1, bf[t], acc[1][t]);
    }
    pa += 32;
    pb += 32;
  }

#pragma unroll
  for (int i = 0; i < 2; ++i) {
#pragma unroll
    for (int t = 0; t < 4; ++t) {
      const int f = f0 + t * 16 + r16;
#pragma unroll
      for (int j = 0; j < 8; ++j) {
        const int m  = m0 + i * 16 + sub * 8 + j;
        const int bb = m >> 11;
        const int n  = m & (ATT_N - 1);
        const _Float16 hv = (_Float16)acc[i][t][j];
        if (f < ATT_E) {
          Qh[((size_t)bb * ATT_N + n) * ATT_E + f] = hv;
        } else if (f < 2 * ATT_E) {
          Kh[((size_t)bb * ATT_N + n) * ATT_E + (f - ATT_E)] = hv;
        } else {
          Vt[((size_t)bb * ATT_E + (f - 2 * ATT_E)) * ATT_N + n] = hv;
        }
      }
    }
  }
}

// ---------------------------------------------------------------------------
// Kernel 2: scores = Q K^T * scale, row softmax -> P (f16)
// One 256-thread block / 16-query stripe. Q stripe (32KB) staged to LDS via
// the Tensor Data Mover; scores stripe (16x2048 f32) lives in LDS.
// ---------------------------------------------------------------------------
__global__ void attn_softmax_kernel(const _Float16* __restrict__ Qh,
                                    const _Float16* __restrict__ Kh,
                                    _Float16* __restrict__ P) {
  extern __shared__ float smem[];              // scores[16*2048] | red[256] | Qlds
  float* red = smem + 16 * ATT_N;
  _Float16* qlds = (_Float16*)(red + 256);     // 16 x 1024 f16 = 32KB

  const int lane = threadIdx.x & 31;
  const int wid  = threadIdx.x >> 5;
  const int sub  = lane >> 4;
  const int r16  = lane & 15;

  const int b  = blockIdx.x >> 7;              // 128 blocks per batch
  const int q0 = (blockIdx.x & 127) * 16;

  const _Float16* Qb = Qh + (size_t)b * ATT_N * ATT_E;
  const _Float16* Kb = Kh + (size_t)b * ATT_N * ATT_E;
  const _Float16* qsrc = Qb + (size_t)q0 * ATT_E;   // 32KB contiguous stripe

#if __has_builtin(__builtin_amdgcn_tensor_load_to_lds) && \
    __has_builtin(__builtin_amdgcn_s_wait_tensorcnt)
  // ---- TDM: DMA the 32KB Q stripe into LDS (1 row of 4096 8-byte elems) ----
  if (wid == 0) {
    const unsigned lds_base = (unsigned)(uintptr_t)(void*)qlds;  // LDS byte offset
    const unsigned long long ga = (unsigned long long)(uintptr_t)qsrc;
    u32x4 g0 = { 1u,                                   // count=1, user mode
                 lds_base,
                 (unsigned)(ga & 0xFFFFFFFFull),       // global_addr[31:0]
                 (unsigned)((ga >> 32) & 0x1FFFFFFull) | (2u << 30) }; // [56:32]|type=2
    i32x8 g1;
    g1[0] = (int)(3u << 16);                 // workgroup_mask=0, data_size=8B
    g1[1] = (int)(4096u << 16);              // tensor_dim0[15:0] in [31:16]
    g1[2] = (int)(1u << 16);                 // tensor_dim0[31:16]=0 | tensor_dim1=1
    g1[3] = (int)(4096u << 16);              // tensor_dim1 hi=0 | tile_dim0=4096
    g1[4] = 1;                               // tile_dim1=1, tile_dim2=0
    g1[5] = 4096;                            // tensor_dim0_stride lo32
    g1[6] = (int)(4096u << 16);              // stride0 hi=0 | tensor_dim1_stride lo16
    g1[7] = 0;                               // tensor_dim1_stride hi
    i32x4 gz4 = { 0, 0, 0, 0 };
    i32x8 gz8 = { 0, 0, 0, 0, 0, 0, 0, 0 };
    __builtin_amdgcn_tensor_load_to_lds(g0, g1, gz4, gz4, gz8, 0);
    __builtin_amdgcn_s_wait_tensorcnt(0);
  }
#else
  // ---- fallback: cooperative copy ----
  {
    const uint4v* s = (const uint4v*)qsrc;
    uint4v* d = (uint4v*)qlds;
    for (int i = threadIdx.x; i < 2048; i += 256) d[i] = s[i];
  }
#endif
  __syncthreads();

  // ---- scores: each wave owns 4 groups of 4 key-tiles (16x64 per group) ----
#pragma unroll 1
  for (int g = wid; g < 32; g += 8) {
    const _Float16* pq = qlds + (size_t)r16 * ATT_E + sub * 16;
    const _Float16* pk = Kb + (size_t)(g * 64 + r16) * ATT_E + sub * 16;
    v8f acc[4] = {};
#pragma unroll 2
    for (int e0 = 0; e0 < ATT_E; e0 += 32) {
      v16h a = make_a(*(const v8h*)(pq), *(const v8h*)(pq + 8));
      v16h kf[4];
#pragma unroll
      for (int t = 0; t < 4; ++t)
        kf[t] = *(const v16h*)(pk + t * 16 * ATT_E);
#pragma unroll
      for (int t = 0; t < 4; ++t) acc[t] = wmma_f16(a, kf[t], acc[t]);
      pq += 32;
      pk += 32;
    }
#pragma unroll
    for (int t = 0; t < 4; ++t)
#pragma unroll
      for (int j = 0; j < 8; ++j)
        smem[(sub * 8 + j) * ATT_N + g * 64 + t * 16 + r16] =
            acc[t][j] * ATT_SCALE;
  }
  __syncthreads();

  // ---- softmax over each of 16 rows (16 threads/row) ----------------------
  const int row  = threadIdx.x >> 4;
  const int part = threadIdx.x & 15;
  const int c0   = part * (ATT_N / 16);

  float m = -3.0e38f;
  for (int c = c0; c < c0 + ATT_N / 16; ++c)
    m = fmaxf(m, smem[row * ATT_N + c]);
  red[row * 16 + part] = m;
  __syncthreads();
  if (part == 0) {
    float mm = red[row * 16];
#pragma unroll
    for (int i = 1; i < 16; ++i) mm = fmaxf(mm, red[row * 16 + i]);
    red[row * 16] = mm;
  }
  __syncthreads();
  const float rowmax = red[row * 16];
  __syncthreads();

  float s = 0.f;
  for (int c = c0; c < c0 + ATT_N / 16; ++c)
    s += __expf(smem[row * ATT_N + c] - rowmax);
  red[row * 16 + part] = s;
  __syncthreads();
  if (part == 0) {
    float ss = 0.f;
#pragma unroll
    for (int i = 0; i < 16; ++i) ss += red[row * 16 + i];
    red[row * 16] = ss;
  }
  __syncthreads();
  const float inv = 1.0f / red[row * 16];

  _Float16* Prow = P + ((size_t)b * ATT_N + q0 + row) * ATT_N;
  for (int j = 0; j < ATT_N / 16; ++j) {
    const int c = part + j * 16;
    Prow[c] = (_Float16)(__expf(smem[row * ATT_N + c] - rowmax) * inv);
  }
}

// ---------------------------------------------------------------------------
// Kernel 3: out = P @ V via Vt (per batch M=2048, N=1024, K=2048), 32x64/wave
// ---------------------------------------------------------------------------
__global__ void pv_kernel(const _Float16* __restrict__ P,
                          const _Float16* __restrict__ Vt,
                          float* __restrict__ out) {
  const int lane = threadIdx.x & 31;
  const int wid  = threadIdx.x >> 5;
  const int sub  = lane >> 4;
  const int r16  = lane & 15;

  const int tile  = blockIdx.x * 4 + wid;      // 4 * 64 * 16 tiles
  const int b     = tile >> 10;
  const int rem   = tile & 1023;
  const int e0    = (rem & 15) * 64;
  const int q0    = (rem >> 4) * 32;

  const _Float16* Pb  = P  + (size_t)b * ATT_N * ATT_N;
  const _Float16* Vtb = Vt + (size_t)b * ATT_E * ATT_N;

  const _Float16* pa = Pb  + (size_t)(q0 + r16) * ATT_N + sub * 16;
  const _Float16* pb = Vtb + (size_t)(e0 + r16) * ATT_N + sub * 16;

  v8f acc[2][4] = {};
#pragma unroll 2
  for (int k0 = 0; k0 < ATT_N; k0 += 32) {
    v16h a0 = make_a(*(const v8h*)(pa),
                     *(const v8h*)(pa + 8));
    v16h a1 = make_a(*(const v8h*)(pa + 16 * ATT_N),
                     *(const v8h*)(pa + 16 * ATT_N + 8));
    v16h vb[4];
#pragma unroll
    for (int t = 0; t < 4; ++t)
      vb[t] = *(const v16h*)(pb + t * 16 * ATT_N);
#pragma unroll
    for (int t = 0; t < 4; ++t) {
      acc[0][t] = wmma_f16(a0, vb[t], acc[0][t]);
      acc[1][t] = wmma_f16(a1, vb[t], acc[1][t]);
    }
    pa += 32;
    pb += 32;
  }

  float* ob = out + (size_t)b * ATT_N * ATT_E;
#pragma unroll
  for (int i = 0; i < 2; ++i)
#pragma unroll
    for (int t = 0; t < 4; ++t)
#pragma unroll
      for (int j = 0; j < 8; ++j)
        ob[(size_t)(q0 + i * 16 + sub * 8 + j) * ATT_E + e0 + t * 16 + r16] =
            acc[i][t][j];
}

// ---------------------------------------------------------------------------
extern "C" void kernel_launch(void* const* d_in, const int* in_sizes, int n_in,
                              void* d_out, int out_size, void* d_ws, size_t ws_size,
                              hipStream_t stream) {
  const float* x = (const float*)d_in[0];       // [4,2048,1024] f32
  const float* W = (const float*)d_in[1];       // [3072,1024]   f32
  float* out = (float*)d_out;

  char* ws = (char*)d_ws;
  _Float16* Qh = (_Float16*)(ws);
  _Float16* Kh = (_Float16*)(ws + ((size_t)16 << 20));
  _Float16* Vt = (_Float16*)(ws + ((size_t)32 << 20));
  _Float16* P  = (_Float16*)(ws + ((size_t)48 << 20));
  // Xh/Wh alias the P region; they are dead before kernel 2 writes P.
  _Float16* Xh = (_Float16*)(ws + ((size_t)48 << 20));
  _Float16* Wh = (_Float16*)(ws + ((size_t)64 << 20));

  // K0: fp32 -> f16 converts (8 elems/thread)
  cvt_f16_kernel<<<(ATT_B * ATT_N * ATT_E) / (256 * 8), 256, 0, stream>>>(x, Xh);
  cvt_f16_kernel<<<(ATT_F * ATT_E) / (256 * 8), 256, 0, stream>>>(W, Wh);

  // K1: 256*48 = 12288 wave tiles / 4 waves per block
  qkv_proj_kernel<<<12288 / 4, 128, 0, stream>>>(Xh, Wh, Qh, Kh, Vt);

  // K2: scores(128KB) + red(1KB) + Q stripe(32KB) dynamic LDS
  const size_t shmem = (size_t)(16 * ATT_N + 256) * sizeof(float) +
                       (size_t)16 * ATT_E * sizeof(_Float16);
  attn_softmax_kernel<<<ATT_B * (ATT_N / 16), 256, shmem, stream>>>(Qh, Kh, P);

  // K3: 4*64*16 = 4096 wave tiles / 4 waves per block
  pv_kernel<<<4096 / 4, 128, 0, stream>>>(P, Vt, out);
}
